// CIN_20126216749634
// MI455X (gfx1250) — matrix operand order repaired and checked by reading, not compiled, not run on hardware
//
#include <hip/hip_runtime.h>

// ---------------------------------------------------------------------------
// CIN (xDeepFM) forward for MI455X / gfx1250.
// bf16 WMMA (v_wmma_f32_16x16x32_bf16) + TDM double-buffered W staging
// (tensor_load_to_lds / s_wait_tensorcnt) + v_pk_mul_bf16 Z-fragment build.
// ---------------------------------------------------------------------------

#define NW0 (128 * 1024)   // W0 elements
#define NW1 (128 * 2048)   // W1 elements
#define NB  2              // batches per block

// W stage: chunks of 64 K-elements x 128 rows, bf16, 4-DWORD pad per row
#define KC      64                    // K elements per chunk
#define SROWU   72                    // staged row stride in u16 (128B + 16B pad)

typedef __attribute__((ext_vector_type(16))) __bf16          v16bf;
typedef __attribute__((ext_vector_type(16))) unsigned short  v16u;
typedef __attribute__((ext_vector_type(8)))  unsigned short  v8u;
typedef __attribute__((ext_vector_type(8)))  float           v8f;
typedef __attribute__((ext_vector_type(4)))  unsigned int    v4u;
typedef __attribute__((ext_vector_type(4)))  unsigned int    u32x4;
typedef __attribute__((ext_vector_type(8)))  int             i32x8;
typedef __attribute__((ext_vector_type(4)))  int             i32x4;

__device__ __forceinline__ unsigned short f2bf(float f) {
  unsigned u = __builtin_bit_cast(unsigned, f);
  u += 0x7FFFu + ((u >> 16) & 1u);          // round-to-nearest-even
  return (unsigned short)(u >> 16);
}

// packed bf16 multiply (VOP3P V_PK_MUL_BF16)
__device__ __forceinline__ unsigned pkmul_bf16(unsigned a, unsigned b) {
  unsigned r;
  asm("v_pk_mul_bf16 %0, %1, %2" : "=v"(r) : "v"(a), "v"(b));
  return r;
}

// A fragment: 16x32 bf16 tile, ISA 16-bit A layout (two 16B chunks per lane)
__device__ __forceinline__ v16bf loadA(const unsigned short* bp) {
  union { v16u v; v8u h[2]; } au;
  au.h[0] = *(const v8u*)bp;         // K 0-7   (lanes>=16: K 8-15)
  au.h[1] = *(const v8u*)(bp + 16);  // K 16-23 (lanes>=16: K 24-31)
  return __builtin_bit_cast(v16bf, au.v);
}

// B fragment: element j = x[8hi+j] * s (j<8), x[16+8hi+j-8] * s (j>=8)  (xv pre-packed bf16 pairs, ss = splat(s))
__device__ __forceinline__ v16bf buildB(const v4u* xv, unsigned ss) {
  union { v16u v; unsigned u[8]; } bu;
#pragma unroll
  for (int j = 0; j < 4; ++j) {
    bu.u[j]     = pkmul_bf16(xv[0][j], ss);
    bu.u[4 + j] = pkmul_bf16(xv[1][j], ss);
  }
  return __builtin_bit_cast(v16bf, bu.v);
}

// generic LDS pointer -> workgroup-relative byte offset (low 32 bits of aperture addr)
__device__ __forceinline__ unsigned lds_off(const void* p) {
  return (unsigned)(unsigned long long)p;
}

// Issue one TDM 2D tile load: 128 rows x KC cols bf16, row-padded in LDS.
// D# per CDNA5 ISA 8.3/8.4: group0 {count=1,type=2,lds,global}, group1 dims.
__device__ __forceinline__ void tdm_load_chunk(unsigned ldsoff,
                                               const unsigned short* gsrc,
                                               unsigned dim0 /* tensor K width */) {
  unsigned long long ga = (unsigned long long)gsrc;
  u32x4 g0;
  g0[0] = 1u;                                   // count=1 (valid), user mode
  g0[1] = ldsoff;                               // lds_addr (bytes)
  g0[2] = (unsigned)ga;                         // global_addr[31:0]
  g0[3] = (unsigned)(ga >> 32) | (2u << 30);    // global_addr[56:32] | type=2
  i32x8 g1;
  g1[0] = (int)((1u << 16)      // data_size = 2 bytes
              | (1u << 20)      // pad_enable
              | (4u << 22)      // pad_interval code 4 -> every 32 DWORDs (one row)
              | (3u << 25));    // pad_amount  code 3 -> 4 DWORDs (16B)
  g1[1] = (int)((dim0 & 0xFFFFu) << 16);                    // tensor_dim0 lo
  g1[2] = (int)((dim0 >> 16) | (128u << 16));               // dim0 hi | tensor_dim1=128
  g1[3] = (int)((unsigned)KC << 16);                        // dim1 hi=0 | tile_dim0=KC
  g1[4] = 128;                                              // tile_dim1=128, tile_dim2=0
  g1[5] = (int)dim0;                                        // tensor_dim0_stride lo
  g1[6] = 0;                                                // stride0 hi | stride1 lo
  g1[7] = 0;                                                // stride1 hi
  i32x4 z4 = {0, 0, 0, 0};
#if __clang_major__ >= 23
  i32x8 z8 = {0, 0, 0, 0, 0, 0, 0, 0};
  __builtin_amdgcn_tensor_load_to_lds(g0, g1, z4, z4, z8, 0);
#else
  __builtin_amdgcn_tensor_load_to_lds(g0, g1, z4, z4, 0);
#endif
}

// ----- one-shot f32 -> bf16 weight conversion into workspace ---------------
__global__ __launch_bounds__(256) void cvt_weights(
    const float* __restrict__ W0, const float* __restrict__ W1,
    unsigned short* __restrict__ wsb) {
  int i = blockIdx.x * blockDim.x + threadIdx.x;     // 0 .. NW1-1
  if (i < NW0) wsb[i] = f2bf(W0[i]);
  if (i < NW1) wsb[NW0 + i] = f2bf(W1[i]);
}

// ----- main kernel: 1024 blocks x 256 threads (8 wave32), NB=2 batches -----
__global__ __launch_bounds__(256) void cin_wmma(
    const float* __restrict__ X,                // (2048, 32, 64) f32
    const unsigned short* __restrict__ Wb,      // bf16: W0 (128x1024), W1 (128x2048)
    const float* __restrict__ b0,
    const float* __restrict__ b1,
    float* __restrict__ Y) {                    // (2048, 192) f32

  __shared__ __align__(16) unsigned short Wstage[2][128 * SROWU]; // 36864B
  __shared__ __align__(16) unsigned short xT[NB][64][32];         // 8192B  [d][m]
  __shared__ __align__(16) unsigned short hT[NB][64][64];         // 16384B [d][hh]
  __shared__ float ylds[NB][192];                                 // 1536B

  const int tid = threadIdx.x;
  for (int e = tid; e < NB * 192; e += 256) ((float*)ylds)[e] = 0.0f;

  // load x (coalesced) and transpose into LDS as bf16
  const long xbase = (long)blockIdx.x * NB * (32 * 64);
  for (int e = tid; e < NB * 32 * 64; e += 256) {
    int bb = e >> 11;
    int r  = e & 2047;
    int m  = r >> 6;
    int d  = r & 63;
    xT[bb][d][m] = f2bf(X[xbase + e]);
  }
  __syncthreads();

  const int wave = tid >> 5;
  const int lane = tid & 31;
  const int a    = wave >> 1;   // M-tile pair {2a, 2a+1}
  const int bq   = wave & 1;    // N-tile pair {2bq, 2bq+1}
  const int hi   = lane >> 4;
  const int col  = lane & 15;
  const int koff = hi ? 8 : 0;

  const unsigned stoff[2] = { lds_off(&Wstage[0][0]), lds_off(&Wstage[1][0]) };
  const v8f vzero = {0.f, 0.f, 0.f, 0.f, 0.f, 0.f, 0.f, 0.f};

  // k-invariant Z factor: x[8hi+j][d] (j<8), x[16+8hi+j-8][d] (j>=8), packed bf16 pairs, live across layers
  v4u xv[NB][2][2];
#pragma unroll
  for (int bb = 0; bb < NB; ++bb)
#pragma unroll
    for (int n = 0; n < 2; ++n) {
      const int d = 16 * (2 * bq + n) + col;
      const v4u* p = (const v4u*)&xT[bb][d][8 * hi];
      xv[bb][n][0] = p[0];
      xv[bb][n][1] = p[2];
    }

  // ============================ layer 0 =====================================
  {
    v8f acc[2][2][NB];
#pragma unroll
    for (int i = 0; i < 2; ++i)
#pragma unroll
      for (int n = 0; n < 2; ++n)
#pragma unroll
        for (int bb = 0; bb < NB; ++bb) acc[i][n][bb] = vzero;

    const unsigned short* W0b = Wb;              // row stride 1024
    const int NC = 1024 / KC;                    // 16 chunks, 2 k-steps each
    if (wave == 0) tdm_load_chunk(stoff[0], W0b, 1024);
    for (int c = 0; c < NC; ++c) {
      if (wave == 0) __builtin_amdgcn_s_wait_tensorcnt(0);
      __syncthreads();                            // chunk c visible to all waves
      if (wave == 0 && c + 1 < NC)
        tdm_load_chunk(stoff[(c + 1) & 1], W0b + (c + 1) * KC, 1024);
      const unsigned short* st = &Wstage[c & 1][0];
#pragma unroll
      for (int kl = 0; kl < 2; ++kl) {
        const int k = 2 * c + kl;                 // k-step 0..31
        v16bf A[2];
#pragma unroll
        for (int i = 0; i < 2; ++i) {
          const int o = 16 * (2 * a + i) + col;
          A[i] = loadA(st + o * SROWU + 32 * kl + koff);
        }
#pragma unroll
        for (int bb = 0; bb < NB; ++bb) {
#pragma unroll
          for (int n = 0; n < 2; ++n) {
            const int d = 16 * (2 * bq + n) + col;
            const unsigned ss = (unsigned)xT[bb][d][k] * 0x10001u;
            v16bf Bf = buildB(xv[bb][n], ss);
#pragma unroll
            for (int i = 0; i < 2; ++i)
              acc[i][n][bb] = __builtin_amdgcn_wmma_f32_16x16x32_bf16(
                  false, A[i], false, Bf, (short)0, acc[i][n][bb], false, false);
          }
        }
      }
      __syncthreads();                            // done reading buf before reuse
    }

    // epilogue: bias + relu; rows <64 -> hT (bf16), rows >=64 -> y sum over d
#pragma unroll
    for (int i = 0; i < 2; ++i) {
      const int mt = 2 * a + i;
#pragma unroll
      for (int bb = 0; bb < NB; ++bb) {
#pragma unroll
        for (int n = 0; n < 2; ++n) {
          const int d = 16 * (2 * bq + n) + col;
          v8f v = acc[i][n][bb];
          if (mt < 4) {
            v8u hv;
#pragma unroll
            for (int r = 0; r < 8; ++r) {
              const int o = 16 * mt + 8 * hi + r;
              float val = v[r] + b0[o];
              hv[r] = f2bf(val > 0.f ? val : 0.f);
            }
            *(v8u*)&hT[bb][d][16 * mt + 8 * hi] = hv;   // ds_store_b128
          } else {
#pragma unroll
            for (int r = 0; r < 8; ++r) {
              const int o = 16 * mt + 8 * hi + r;
              float val = v[r] + b0[o];
              val = val > 0.f ? val : 0.f;
              atomicAdd(&ylds[bb][o - 64], val);        // ds_add_f32
            }
          }
        }
      }
    }
  }
  __syncthreads();

  // ============================ layer 1 =====================================
  {
    v8f acc[2][2][NB];
#pragma unroll
    for (int i = 0; i < 2; ++i)
#pragma unroll
      for (int n = 0; n < 2; ++n)
#pragma unroll
        for (int bb = 0; bb < NB; ++bb) acc[i][n][bb] = vzero;

    const unsigned short* W1b = Wb + NW0;        // row stride 2048
    const int NC = 2048 / KC;                    // 32 chunks
    if (wave == 0) tdm_load_chunk(stoff[0], W1b, 2048);
    for (int c = 0; c < NC; ++c) {
      if (wave == 0) __builtin_amdgcn_s_wait_tensorcnt(0);
      __syncthreads();
      if (wave == 0 && c + 1 < NC)
        tdm_load_chunk(stoff[(c + 1) & 1], W1b + (c + 1) * KC, 2048);
      const unsigned short* st = &Wstage[c & 1][0];
#pragma unroll
      for (int kl = 0; kl < 2; ++kl) {
        const int k = 2 * c + kl;                 // k-step 0..63
        v16bf A[2];
#pragma unroll
        for (int i = 0; i < 2; ++i) {
          const int o = 16 * (2 * a + i) + col;
          A[i] = loadA(st + o * SROWU + 32 * kl + koff);
        }
#pragma unroll
        for (int bb = 0; bb < NB; ++bb) {
#pragma unroll
          for (int n = 0; n < 2; ++n) {
            const int d = 16 * (2 * bq + n) + col;
            const unsigned ss = (unsigned)hT[bb][d][k] * 0x10001u;
            v16bf Bf = buildB(xv[bb][n], ss);
#pragma unroll
            for (int i = 0; i < 2; ++i)
              acc[i][n][bb] = __builtin_amdgcn_wmma_f32_16x16x32_bf16(
                  false, A[i], false, Bf, (short)0, acc[i][n][bb], false, false);
          }
        }
      }
      __syncthreads();
    }

    // epilogue: bias + relu + sum over d into y[64 + o]
#pragma unroll
    for (int i = 0; i < 2; ++i) {
#pragma unroll
      for (int bb = 0; bb < NB; ++bb) {
#pragma unroll
        for (int n = 0; n < 2; ++n) {
          v8f v = acc[i][n][bb];
#pragma unroll
          for (int r = 0; r < 8; ++r) {
            const int o = 16 * (2 * a + i) + 8 * hi + r;
            float val = v[r] + b1[o];
            val = val > 0.f ? val : 0.f;
            atomicAdd(&ylds[bb][64 + o], val);
          }
        }
      }
    }
  }
  __syncthreads();

  // write y: (block batches) x 192, coalesced
  for (int e = tid; e < NB * 192; e += 256) {
    const int bb = e / 192;
    const int i  = e % 192;
    Y[((long)blockIdx.x * NB + bb) * 192 + i] = ylds[bb][i];
  }
}

// ---------------------------------------------------------------------------
extern "C" void kernel_launch(void* const* d_in, const int* in_sizes, int n_in,
                              void* d_out, int out_size, void* d_ws, size_t ws_size,
                              hipStream_t stream) {
  const float* X  = (const float*)d_in[0];   // (2048, 32, 64)
  const float* W0 = (const float*)d_in[1];   // (128, 1024)
  const float* b0 = (const float*)d_in[2];   // (128,)
  const float* W1 = (const float*)d_in[3];   // (128, 2048)
  const float* b1 = (const float*)d_in[4];   // (128,)
  float* Y = (float*)d_out;                  // (2048, 192)
  unsigned short* wsb = (unsigned short*)d_ws;  // bf16 W0 | bf16 W1 (768 KB)

  cvt_weights<<<NW1 / 256, 256, 0, stream>>>(W0, W1, wsb);
  cin_wmma<<<2048 / NB, 256, 0, stream>>>(X, wsb, b0, b1, Y);
}
